// Baseline_42236708389039
// MI455X (gfx1250) — compile-verified, matrix operand order checked by reading, not run on hardware
//
#include <hip/hip_runtime.h>

// ---------------------------------------------------------------------------
// GAT block for MI455X (gfx1250): all 4 contractions via v_wmma_f32_16x16x32_bf16
// with 32x64 register blocking per wave (2 A-frags x 4 B-frags, 8 accumulators)
// B=32, N=256, F=300 (pad 320), H=8, O=256, HO=2048, OUT=512
// ---------------------------------------------------------------------------

typedef __attribute__((ext_vector_type(16))) __bf16        v16bf;
typedef __attribute__((ext_vector_type(8)))  float         v8f;
typedef __attribute__((ext_vector_type(4)))  unsigned int  v4u;

union FragBF { v16bf bf; v4u q[2]; };

__device__ __forceinline__ unsigned short f2bf(float x) {
    unsigned u = __float_as_uint(x);
    unsigned r = u + 0x7FFFu + ((u >> 16) & 1u);   // round-to-nearest-even
    return (unsigned short)(r >> 16);
}
__device__ __forceinline__ float bf2f(unsigned short b) {
    return __uint_as_float(((unsigned)b) << 16);
}

// One wave computes a (16*MT) x (16*NT) f32 tile:  C += A(16MTxK) * B(Kx16NT)
// A: row-major bf16, lda elements (K contiguous).  Bt: per output column, K
// contiguous (i.e. B stored transposed), ldb elements.  K % 32 == 0,
// lda/ldb % 8 == 0 -> every fragment load is a 16B-aligned b128.
// 16-bit A frag layout (ISA 7.12.2): lanes 0-15 row M=lane hold K k0+0..7
// (VGPR0-3) and k0+16..23 (VGPR4-7); lanes 16-31 hold the +8 K-halves.
// B fragment uses the symmetric per-column layout.
template<int MT, int NT>
__device__ __forceinline__ void wmma_bf16_block(const unsigned short* __restrict__ A,
                                                int lda,
                                                const unsigned short* __restrict__ Bt,
                                                int ldb, int K, v8f (&acc)[MT][NT]) {
    const int lane = threadIdx.x & 31;
    const int half = lane >> 4;          // 0: lanes 0-15, 1: lanes 16-31
    const int r    = lane & 15;
    const unsigned short* arow[MT];
    const unsigned short* bcol[NT];
#pragma unroll
    for (int i = 0; i < MT; ++i) arow[i] = A  + (size_t)(r + 16 * i) * lda;
#pragma unroll
    for (int j = 0; j < NT; ++j) bcol[j] = Bt + (size_t)(r + 16 * j) * ldb;

    for (int k0 = 0; k0 < K; k0 += 32) {
        const int o1 = k0 + half * 8;
        FragBF fa[MT], fb[NT];
#pragma unroll
        for (int i = 0; i < MT; ++i) {
            fa[i].q[0] = *(const v4u*)(arow[i] + o1);
            fa[i].q[1] = *(const v4u*)(arow[i] + o1 + 16);
        }
#pragma unroll
        for (int j = 0; j < NT; ++j) {
            fb[j].q[0] = *(const v4u*)(bcol[j] + o1);
            fb[j].q[1] = *(const v4u*)(bcol[j] + o1 + 16);
        }
        if (k0 + 32 < K) {               // -> global_prefetch_b8 (near caches)
            __builtin_prefetch(arow[0] + o1 + 32, 0, 3);
            __builtin_prefetch(bcol[0] + o1 + 32, 0, 3);
        }
#pragma unroll
        for (int i = 0; i < MT; ++i)
#pragma unroll
            for (int j = 0; j < NT; ++j)
                acc[i][j] = __builtin_amdgcn_wmma_f32_16x16x32_bf16(
                                false, fa[i].bf, false, fb[j].bf,
                                (short)0, acc[i][j], false, false);
    }
}

// ------------------------- prep kernels ------------------------------------

// x_bf16[(b*256+n)*320 + f] = embed[fea[b,n], f]  (zero-padded f>=300)
__global__ void k_embed_gather(const int* __restrict__ fea,
                               const float* __restrict__ embed,
                               unsigned short* __restrict__ xb) {
    const int bn = blockIdx.x, f = threadIdx.x;          // 8192 x 320
    const int tok = fea[bn];
    float v = (f < 300) ? embed[(size_t)tok * 300 + f] : 0.f;
    xb[(size_t)bn * 320 + f] = f2bf(v);
}

// w1t[(h*256+o)*320 + f] = W_heads[h, f, o]
__global__ void k_w1_transpose(const float* __restrict__ W,
                               unsigned short* __restrict__ wt) {
    const int ho = blockIdx.x, f = threadIdx.x;          // 2048 x 320
    const int h = ho >> 8, o = ho & 255;
    float v = (f < 300) ? W[((size_t)h * 300 + f) * 256 + o] : 0.f;
    wt[(size_t)ho * 320 + f] = f2bf(v);
}

// w2t[o*2048 + k] = W_out[k, o]
__global__ void k_w2_transpose(const float* __restrict__ W,
                               unsigned short* __restrict__ wt) {
    const int o = blockIdx.x;                            // 512 x 256
    for (int k = threadIdx.x; k < 2048; k += blockDim.x)
        wt[(size_t)o * 2048 + k] = f2bf(W[(size_t)k * 512 + o]);
}

// ------------------------- layer-1 GEMM ------------------------------------

// Wh^T (bf16): whT[((b*8+h)*256 + o)*256 + n] = (x @ W_h)[b,n,o]
__global__ void k_gemm1(const unsigned short* __restrict__ xb,
                        const unsigned short* __restrict__ w1t,
                        unsigned short* __restrict__ whT) {
    const int m0 = blockIdx.x * 32;                      // row in (b*256+n)
    const int n0 = blockIdx.y * 64;                      // o tile
    const int h  = blockIdx.z;
    v8f acc[2][4] = {};
    wmma_bf16_block<2, 4>(xb + (size_t)m0 * 320, 320,
                          w1t + ((size_t)h * 256 + n0) * 320, 320, 320, acc);
    const int lane = threadIdx.x & 31, half = lane >> 4;
    const int b = m0 >> 8, nb = m0 & 255;                // 32-tiles never cross b
#pragma unroll
    for (int i = 0; i < 2; ++i)
#pragma unroll
        for (int j = 0; j < 4; ++j) {
            const int col = n0 + j * 16 + (lane & 15);
            const size_t cbase =
                ((size_t)(b * 8 + h) * 256 + col) * 256 + nb + i * 16 + half * 8;
#pragma unroll
            for (int r = 0; r < 8; ++r)
                whT[cbase + r] = f2bf(acc[i][j][r]);
        }
}

// ------------------------- edge scores -------------------------------------

// e1[idx]=Wh.a[:O], e2[idx]=Wh.a[O:], idx=(b*8+h)*256+n   (layer 1)
__global__ void k_edge_scores1(const unsigned short* __restrict__ whT,
                               const float* __restrict__ a_heads,
                               float* __restrict__ e1, float* __restrict__ e2) {
    const int idx = blockIdx.x;                          // 65536 x 32
    const int n = idx & 255, bh = idx >> 8, h = bh & 7;
    const int lane = threadIdx.x;
    float s1 = 0.f, s2 = 0.f;
    for (int o = lane; o < 256; o += 32) {
        float w = bf2f(whT[((size_t)bh * 256 + o) * 256 + n]);
        s1 += w * a_heads[h * 512 + o];
        s2 += w * a_heads[h * 512 + 256 + o];
    }
#pragma unroll
    for (int off = 16; off; off >>= 1) {
        s1 += __shfl_xor(s1, off, 32);
        s2 += __shfl_xor(s2, off, 32);
    }
    if (lane == 0) { e1[idx] = s1; e2[idx] = s2; }
}

// layer 2: idx = b*256+n, reduce over 512
__global__ void k_edge_scores2(const unsigned short* __restrict__ wh2T,
                               const float* __restrict__ a_out,
                               float* __restrict__ e1, float* __restrict__ e2) {
    const int idx = blockIdx.x;                          // 8192 x 32
    const int n = idx & 255, b = idx >> 8;
    const int lane = threadIdx.x;
    float s1 = 0.f, s2 = 0.f;
    for (int o = lane; o < 512; o += 32) {
        float w = bf2f(wh2T[((size_t)(b * 512 + o)) * 256 + n]);
        s1 += w * a_out[o];
        s2 += w * a_out[512 + o];
    }
#pragma unroll
    for (int off = 16; off; off >>= 1) {
        s1 += __shfl_xor(s1, off, 32);
        s2 += __shfl_xor(s2, off, 32);
    }
    if (lane == 0) { e1[idx] = s1; e2[idx] = s2; }
}

// ------------------------- masked softmax ----------------------------------

// idx = (row over heads*nodes); hshift=3 for layer1 (8 heads), 0 for layer2.
__global__ void k_softmax_row(const float* __restrict__ e1,
                              const float* __restrict__ e2,
                              const int* __restrict__ adj, int hshift,
                              unsigned short* __restrict__ attn) {
    __shared__ float red[256];
    const int idx = blockIdx.x;
    const int n = idx & 255, bh = idx >> 8, b = bh >> hshift;
    const int m = threadIdx.x;
    float x = e1[idx] + e2[bh * 256 + m];
    x = (x > 0.f) ? x : 0.2f * x;                        // LeakyReLU(0.2)
    if (adj[((size_t)(b * 256 + n)) * 256 + m] <= 0) x = -9e15f;
    red[m] = x; __syncthreads();
    for (int s = 128; s; s >>= 1) {
        if (m < s) red[m] = fmaxf(red[m], red[m + s]);
        __syncthreads();
    }
    const float mx = red[0]; __syncthreads();
    const float ex = __expf(x - mx);
    red[m] = ex; __syncthreads();
    for (int s = 128; s; s >>= 1) {
        if (m < s) red[m] += red[m + s];
        __syncthreads();
    }
    attn[(size_t)idx * 256 + m] = f2bf(ex / red[0]);
}

// ------------------------- attention * V (layer 1) -------------------------

// h_bf16[(b*256+n)*2048 + h*256 + o] = bf16( elu(attn @ Wh) * non_pad )
__global__ void k_av1_elu_concat(const unsigned short* __restrict__ attn,
                                 const unsigned short* __restrict__ whT,
                                 const float* __restrict__ non_pad,
                                 unsigned short* __restrict__ hb) {
    const int n0 = blockIdx.x * 32, o0 = blockIdx.y * 64;
    const int bh = blockIdx.z, b = bh >> 3, h = bh & 7;
    v8f acc[2][4] = {};
    wmma_bf16_block<2, 4>(attn + (size_t)bh * 65536 + (size_t)n0 * 256, 256,
                          whT  + (size_t)bh * 65536 + (size_t)o0 * 256, 256,
                          256, acc);
    const int lane = threadIdx.x & 31, half = lane >> 4;
#pragma unroll
    for (int i = 0; i < 2; ++i)
#pragma unroll
        for (int j = 0; j < 4; ++j) {
            const int o = o0 + j * 16 + (lane & 15);
#pragma unroll
            for (int r = 0; r < 8; ++r) {
                const int n = n0 + i * 16 + r + half * 8;
                float v = acc[i][j][r];
                v = (v > 0.f) ? v : (__expf(v) - 1.f);   // ELU
                v *= non_pad[b * 256 + n];
                hb[((size_t)(b * 256 + n)) * 2048 + h * 256 + o] = f2bf(v);
            }
        }
}

// ------------------------- layer-2 GEMM ------------------------------------

// wh2T[(b*512+o)*256 + n] = (h @ W_out)[b,n,o]
__global__ void k_gemm2(const unsigned short* __restrict__ hb,
                        const unsigned short* __restrict__ w2t,
                        unsigned short* __restrict__ wh2T) {
    const int m0 = blockIdx.x * 32;                      // (b*256+n)
    const int o0 = blockIdx.y * 64;                      // out feature
    v8f acc[2][4] = {};
    wmma_bf16_block<2, 4>(hb + (size_t)m0 * 2048, 2048,
                          w2t + (size_t)o0 * 2048, 2048, 2048, acc);
    const int lane = threadIdx.x & 31, half = lane >> 4;
    const int b = m0 >> 8, nb = m0 & 255;
#pragma unroll
    for (int i = 0; i < 2; ++i)
#pragma unroll
        for (int j = 0; j < 4; ++j) {
            const int o = o0 + j * 16 + (lane & 15);
            const size_t cbase =
                ((size_t)(b * 512 + o)) * 256 + nb + i * 16 + half * 8;
#pragma unroll
            for (int r = 0; r < 8; ++r)
                wh2T[cbase + r] = f2bf(acc[i][j][r]);
        }
}

// ------------------------- attention * V (layer 2, final) ------------------

__global__ void k_av2_final(const unsigned short* __restrict__ attn2,
                            const unsigned short* __restrict__ wh2T,
                            const float* __restrict__ non_pad,
                            float* __restrict__ out) {
    const int n0 = blockIdx.x * 32, o0 = blockIdx.y * 64, b = blockIdx.z;
    v8f acc[2][4] = {};
    wmma_bf16_block<2, 4>(attn2 + ((size_t)b * 256 + n0) * 256, 256,
                          wh2T  + ((size_t)b * 512 + o0) * 256, 256, 256, acc);
    const int lane = threadIdx.x & 31, half = lane >> 4;
#pragma unroll
    for (int i = 0; i < 2; ++i)
#pragma unroll
        for (int j = 0; j < 4; ++j) {
            const int o = o0 + j * 16 + (lane & 15);
#pragma unroll
            for (int r = 0; r < 8; ++r) {
                const int n = n0 + i * 16 + r + half * 8;
                float v = acc[i][j][r] * non_pad[b * 256 + n];
                v = (v > 0.f) ? v : (__expf(v) - 1.f);   // ELU
                out[((size_t)(b * 256 + n)) * 512 + o] = v;
            }
        }
}

// ---------------------------------------------------------------------------

extern "C" void kernel_launch(void* const* d_in, const int* in_sizes, int n_in,
                              void* d_out, int out_size, void* d_ws, size_t ws_size,
                              hipStream_t stream) {
    const int*   fea      = (const int*)  d_in[0];   // (32,256)
    const int*   adj      = (const int*)  d_in[1];   // (32,256,256)
    const float* non_pad  = (const float*)d_in[2];   // (32,256,1)
    const float* embed    = (const float*)d_in[3];   // (200,300)
    const float* W_heads  = (const float*)d_in[4];   // (8,300,256)
    const float* a_heads  = (const float*)d_in[5];   // (8,512)
    const float* W_out    = (const float*)d_in[6];   // (2048,512)
    const float* a_out    = (const float*)d_in[7];   // (1024)
    float* out = (float*)d_out;                      // (32,256,512) f32

    // scratch carve-out (~123 MB), 256B aligned
    char* ws = (char*)d_ws;
    size_t off = 0;
    auto carve = [&](size_t bytes) {
        char* p = ws + off;
        off = (off + bytes + 255) & ~(size_t)255;
        return p;
    };
    unsigned short* xb    = (unsigned short*)carve((size_t)8192 * 320 * 2);
    unsigned short* w1t   = (unsigned short*)carve((size_t)2048 * 320 * 2);
    unsigned short* w2t   = (unsigned short*)carve((size_t)512 * 2048 * 2);
    unsigned short* whT   = (unsigned short*)carve((size_t)32 * 8 * 256 * 256 * 2);
    float*          e1    = (float*)carve((size_t)65536 * 4);
    float*          e2    = (float*)carve((size_t)65536 * 4);
    unsigned short* attn1 = (unsigned short*)carve((size_t)32 * 8 * 256 * 256 * 2);
    unsigned short* hb    = (unsigned short*)carve((size_t)8192 * 2048 * 2);
    unsigned short* wh2T  = (unsigned short*)carve((size_t)32 * 512 * 256 * 2);
    float*          e1o   = (float*)carve((size_t)8192 * 4);
    float*          e2o   = (float*)carve((size_t)8192 * 4);
    unsigned short* attn2 = (unsigned short*)carve((size_t)32 * 256 * 256 * 2);
    (void)ws_size; (void)n_in; (void)in_sizes; (void)out_size;

    // prep
    k_embed_gather<<<dim3(8192), dim3(320), 0, stream>>>(fea, embed, xb);
    k_w1_transpose<<<dim3(2048), dim3(320), 0, stream>>>(W_heads, w1t);
    k_w2_transpose<<<dim3(512), dim3(256), 0, stream>>>(W_out, w2t);

    // layer 1
    k_gemm1<<<dim3(256, 4, 8), dim3(32), 0, stream>>>(xb, w1t, whT);
    k_edge_scores1<<<dim3(65536), dim3(32), 0, stream>>>(whT, a_heads, e1, e2);
    k_softmax_row<<<dim3(65536), dim3(256), 0, stream>>>(e1, e2, adj, 3, attn1);
    k_av1_elu_concat<<<dim3(8, 4, 256), dim3(32), 0, stream>>>(attn1, whT, non_pad, hb);

    // layer 2
    k_gemm2<<<dim3(256, 8), dim3(32), 0, stream>>>(hb, w2t, wh2T);
    k_edge_scores2<<<dim3(8192), dim3(32), 0, stream>>>(wh2T, a_out, e1o, e2o);
    k_softmax_row<<<dim3(8192), dim3(256), 0, stream>>>(e1o, e2o, adj, 0, attn2);
    k_av2_final<<<dim3(8, 8, 32), dim3(32), 0, stream>>>(attn2, wh2T, non_pad, out);
}
